// LanguagePuzzleVAESmall_27908697489602
// MI455X (gfx1250) — matrix operand order repaired and checked by reading, not compiled, not run on hardware
//
#include <hip/hip_runtime.h>
#include <hip/hip_bf16.h>

typedef __attribute__((ext_vector_type(16))) _Float16 v16h;
typedef __attribute__((ext_vector_type(8)))  float    v8f;
typedef __attribute__((ext_vector_type(4)))  float    v4f;

#define B_   2048
#define L_   128
#define V_   128
#define E_   32
#define D_   64
#define NC_  10

// ---- d_out layout (floats, concatenated in return order) ----
// char_logits [B,L,V], puzzles [B,30,30], features [B,30,30,D],
// quantized_st [B,30,30,D], vq_loss [1]
#define OUT_LOGITS   ((size_t)0)
#define OUT_PUZZLES  ((size_t)B_ * L_ * V_)                       // 33554432
#define OUT_FEATURES (OUT_PUZZLES + (size_t)B_ * 30 * 30)         // +1843200
#define OUT_QUANT    (OUT_FEATURES + (size_t)B_ * 30 * 30 * D_)   // +117964800
#define OUT_LOSS     (OUT_QUANT + (size_t)B_ * 30 * 30 * D_)

// ---- workspace layout (floats) ----
#define WS_ENC  ((size_t)0)                         // [B,30,64] time-major
#define WS_F    (WS_ENC + (size_t)B_ * 30 * D_)     // [B,64,30] channel-major
#define WS_DEC  (WS_F + (size_t)B_ * D_ * 30)       // [B,128,32] token-major

// ============================================================
// Kernel 1: fused embedding gather + 3 encoder convs, one block per batch elem
// ============================================================
__global__ void encoder_kernel(const int* __restrict__ idx,
                               const float* __restrict__ emb,
                               const float* __restrict__ w1, const float* __restrict__ b1,
                               const float* __restrict__ w2, const float* __restrict__ b2,
                               const float* __restrict__ w3, const float* __restrict__ b3,
                               float* __restrict__ ws) {
    __shared__ float xs[E_][130];   // halo: xs[c][p+1] = x[c][p], p in [-1,128]
    __shared__ float h1[D_][66];    // halo: h1[c][t+1]
    __shared__ float h2[D_][32];
    const int b   = blockIdx.x;
    const int tid = threadIdx.x;

    // gather embedding: x[c][l] = emb[idx[b,l], c]
    for (int i = tid; i < L_ * E_; i += 256) {
        int l = i >> 5, c = i & 31;
        int t = idx[b * L_ + l];
        xs[c][l + 1] = emb[t * E_ + c];
    }
    for (int i = tid; i < E_; i += 256) { xs[i][0] = 0.f; xs[i][129] = 0.f; }
    __syncthreads();

    // conv1: 32->64 ch, k3 s2 p1, len 128->64, relu
    for (int i = tid; i < 64 * 64; i += 256) {
        int oc = i >> 6, ol = i & 63;
        float acc = b1[oc];
        for (int ic = 0; ic < E_; ++ic) {
            const float* w = w1 + (oc * E_ + ic) * 3;
            acc += xs[ic][2 * ol + 0] * w[0] + xs[ic][2 * ol + 1] * w[1] + xs[ic][2 * ol + 2] * w[2];
        }
        h1[oc][ol + 1] = acc > 0.f ? acc : 0.f;
    }
    for (int i = tid; i < 64; i += 256) { h1[i][0] = 0.f; h1[i][65] = 0.f; }
    __syncthreads();

    // conv2: 64->64 ch, k3 s2 p1, len 64->32, relu
    for (int i = tid; i < 64 * 32; i += 256) {
        int oc = i >> 5, ol = i & 31;
        float acc = b2[oc];
        for (int ic = 0; ic < 64; ++ic) {
            const float* w = w2 + (oc * 64 + ic) * 3;
            acc += h1[ic][2 * ol + 0] * w[0] + h1[ic][2 * ol + 1] * w[1] + h1[ic][2 * ol + 2] * w[2];
        }
        h2[oc][ol] = acc > 0.f ? acc : 0.f;
    }
    __syncthreads();

    // conv3: 64->64 ch, k3 s1 p0, len 32->30, no relu; write [B,30,64]
    for (int i = tid; i < 64 * 30; i += 256) {
        int oc = i / 30, t = i % 30;
        float acc = b3[oc];
        for (int ic = 0; ic < 64; ++ic) {
            const float* w = w3 + (oc * 64 + ic) * 3;
            acc += h2[ic][t] * w[0] + h2[ic][t + 1] * w[1] + h2[ic][t + 2] * w[2];
        }
        ws[WS_ENC + ((size_t)b * 30 + t) * D_ + oc] = acc;
    }
}

// ============================================================
// Kernel 2: VQ (argmin over 10 codes) + broadcast stores.
// This carries ~90% of all output bytes -> 128-bit non-temporal stores.
// one block (64 threads) per (b, t)
// ============================================================
__global__ void vq_kernel(const float* __restrict__ ws_enc,
                          const float* __restrict__ codebook,
                          float* __restrict__ out, float* __restrict__ ws_f,
                          float loss_scale) {
    __shared__ __align__(16) float e[D_];
    __shared__ __align__(16) float q[D_];
    __shared__ float dist[NC_];
    __shared__ int   sh_code;
    const int blk = blockIdx.x;
    const int b = blk / 30, t = blk % 30;
    const int tid = threadIdx.x;

    e[tid] = ws_enc[((size_t)b * 30 + t) * D_ + tid];
    __syncthreads();

    if (tid < NC_) {
        float d2 = 0.f;
        const float* cb = codebook + tid * D_;
        for (int d = 0; d < D_; ++d) { float df = e[d] - cb[d]; d2 += df * df; }
        dist[tid] = d2;
    }
    __syncthreads();
    if (tid == 0) {
        int best = 0; float bd = dist[0];
        for (int c = 1; c < NC_; ++c) if (dist[c] < bd) { bd = dist[c]; best = c; }
        sh_code = best;
        atomicAdd(out + OUT_LOSS, bd * loss_scale);
    }
    __syncthreads();
    const int code = sh_code;
    const float qv = codebook[code * D_ + tid];
    q[tid] = qv;

    // decoder input f = mean over broadcast axis = quantized row; store [B,64ch,30]
    ws_f[((size_t)b * D_ + tid) * 30 + t] = qv;

    // puzzles [B,30,30] (float cast of argmin) — same code for all j
    float* puz = out + OUT_PUZZLES + ((size_t)b * 30 + t) * 30;
    if (tid < 30) puz[tid] = (float)code;
    __syncthreads();

    // features / quantized_st broadcast: 30 contiguous rows x 64 floats.
    // 64 threads = 16 vec4 columns x 4 rows per pass; NT hint (never re-read).
    const int c4 = tid & 15;   // vec4 column
    const int r4 = tid >> 4;   // row within group of 4
    const v4f ev4 = ((const v4f*)e)[c4];
    const v4f qv4 = ((const v4f*)q)[c4];
    v4f* feat4 = (v4f*)(out + OUT_FEATURES + (((size_t)b * 30 + t) * 30) * D_);
    v4f* quan4 = (v4f*)(out + OUT_QUANT    + (((size_t)b * 30 + t) * 30) * D_);
    #pragma unroll
    for (int jb = 0; jb < 7; ++jb) {
        int j = jb * 4 + r4;
        __builtin_nontemporal_store(ev4, feat4 + j * 16 + c4);
        __builtin_nontemporal_store(qv4, quan4 + j * 16 + c4);
    }
    if (r4 < 2) {
        int j = 28 + r4;
        __builtin_nontemporal_store(ev4, feat4 + j * 16 + c4);
        __builtin_nontemporal_store(qv4, quan4 + j * 16 + c4);
    }
}

// ============================================================
// Kernel 3: fused 3 transposed convs, one block per batch elem
// ============================================================
__global__ void decoder_kernel(const float* __restrict__ ws_f,
                               const float* __restrict__ w1, const float* __restrict__ b1,
                               const float* __restrict__ w2, const float* __restrict__ b2,
                               const float* __restrict__ w3, const float* __restrict__ b3,
                               float* __restrict__ ws_dec) {
    __shared__ float f [D_][30];
    __shared__ float g1[64][32];
    __shared__ float g2[64][64];
    const int b   = blockIdx.x;
    const int tid = threadIdx.x;

    for (int i = tid; i < D_ * 30; i += 256) {
        int c = i / 30, t = i % 30;
        f[c][t] = ws_f[((size_t)b * D_ + c) * 30 + t];
    }
    __syncthreads();

    // convT1: k3 s1 p0, 64->64, len 30->32, relu.  y[oc,t]=sum f[ic,t-k]*w[ic,oc,k]
    for (int i = tid; i < 64 * 32; i += 256) {
        int oc = i >> 5, t = i & 31;
        float acc = b1[oc];
        for (int ic = 0; ic < 64; ++ic) {
            const float* w = w1 + (ic * 64 + oc) * 3;
            #pragma unroll
            for (int k = 0; k < 3; ++k) {
                int s = t - k;
                if (s >= 0 && s < 30) acc += f[ic][s] * w[k];
            }
        }
        g1[oc][t] = acc > 0.f ? acc : 0.f;
    }
    __syncthreads();

    // convT2: k4 s2 p1, 64->64, len 32->64, relu.  k = t+1-2s
    for (int i = tid; i < 64 * 64; i += 256) {
        int oc = i >> 6, t = i & 63;
        float acc = b2[oc];
        for (int ic = 0; ic < 64; ++ic) {
            const float* w = w2 + (ic * 64 + oc) * 4;
            #pragma unroll
            for (int k = 0; k < 4; ++k) {
                int num = t + 1 - k;
                if (num >= 0 && (num & 1) == 0) {
                    int s = num >> 1;
                    if (s < 32) acc += g1[ic][s] * w[k];
                }
            }
        }
        g2[oc][t] = acc > 0.f ? acc : 0.f;
    }
    __syncthreads();

    // convT3: k4 s2 p1, 64->32, len 64->128, no relu; write [B,L,E] token-major
    for (int i = tid; i < L_ * E_; i += 256) {
        int l = i >> 5, e = i & 31;
        float acc = b3[e];
        for (int ic = 0; ic < 64; ++ic) {
            const float* w = w3 + (ic * E_ + e) * 4;
            #pragma unroll
            for (int k = 0; k < 4; ++k) {
                int num = l + 1 - k;
                if (num >= 0 && (num & 1) == 0) {
                    int s = num >> 1;
                    if (s < 64) acc += g2[ic][s] * w[k];
                }
            }
        }
        ws_dec[((size_t)b * L_ + l) * E_ + e] = acc;
    }
}

// ============================================================
// Kernel 4: logits GEMM  [B*L,32] x [32,128] via v_wmma_f32_16x16x32_f16
// One 16x16 output tile per wave (K=32 == one WMMA). 8 waves/block cover N=128.
// ============================================================
__global__ void logits_wmma_kernel(const float* __restrict__ dec,
                                   const float* __restrict__ out_w,
                                   const float* __restrict__ out_b,
                                   float* __restrict__ logits) {
    const int tid  = threadIdx.x;
    const int wave = tid >> 5;          // 0..7 -> vocab column tile
    const int lane = tid & 31;          // wave32
    const int row0 = blockIdx.x * 16;   // token tile
    const int col0 = wave * 16;
    const int hi   = lane >> 4;         // 0 for lanes 0-15, 1 for 16-31
    const int l16  = lane & 15;

    // A (16x32 f16): row m = lane&15; half h -> K = (h<8 ? h : h+8) + 8*hi
    const float* arow = dec + (size_t)(row0 + l16) * E_;
    v16h a;
    #pragma unroll
    for (int h = 0; h < 16; ++h) {
        int k = (h < 8 ? h : h + 8) + (hi << 3);
        a[h] = (_Float16)arow[k];
    }
    // B (32x16 f16): col n = lane&15; half h -> K = (h<8 ? h : h+8) + 8*hi.  B[k][n] = out_w[n*32+k]
    const float* brow = out_w + (size_t)(col0 + l16) * E_;
    v16h bm;
    #pragma unroll
    for (int h = 0; h < 16; ++h) {
        int k = (h < 8 ? h : h + 8) + (hi << 3);
        bm[h] = (_Float16)brow[k];
    }

    v8f c = {};
    c = __builtin_amdgcn_wmma_f32_16x16x32_f16(false, a, false, bm,
                                               (short)0, c, false, false);

    const float bias = out_b[col0 + l16];
    #pragma unroll
    for (int r = 0; r < 8; ++r) {
        int m = r + (hi << 3);
        logits[(size_t)(row0 + m) * V_ + col0 + l16] = c[r] + bias;
    }
}

// ============================================================
extern "C" void kernel_launch(void* const* d_in, const int* in_sizes, int n_in,
                              void* d_out, int out_size, void* d_ws, size_t ws_size,
                              hipStream_t stream) {
    (void)in_sizes; (void)n_in; (void)out_size; (void)ws_size;
    const int*   text_idx = (const int*)  d_in[0];
    const float* emb      = (const float*)d_in[1];
    const float* enc_w1   = (const float*)d_in[2];
    const float* enc_b1   = (const float*)d_in[3];
    const float* enc_w2   = (const float*)d_in[4];
    const float* enc_b2   = (const float*)d_in[5];
    const float* enc_w3   = (const float*)d_in[6];
    const float* enc_b3   = (const float*)d_in[7];
    const float* codebook = (const float*)d_in[8];
    const float* dec_w1   = (const float*)d_in[9];
    const float* dec_b1   = (const float*)d_in[10];
    const float* dec_w2   = (const float*)d_in[11];
    const float* dec_b2   = (const float*)d_in[12];
    const float* dec_w3   = (const float*)d_in[13];
    const float* dec_b3   = (const float*)d_in[14];
    const float* out_w    = (const float*)d_in[15];
    const float* out_b    = (const float*)d_in[16];

    float* out = (float*)d_out;
    float* ws  = (float*)d_ws;

    // zero the loss accumulator (graph-capture safe)
    (void)hipMemsetAsync(out + OUT_LOSS, 0, sizeof(float), stream);

    encoder_kernel<<<dim3(B_), dim3(256), 0, stream>>>(
        text_idx, emb, enc_w1, enc_b1, enc_w2, enc_b2, enc_w3, enc_b3, ws);

    // vq_loss = 1.25 * sum(d2min) / (B*30*64)   (embedding == commitment msd;
    // broadcast axis cancels in the mean)
    const float loss_scale = 1.25f / (float)((size_t)B_ * 30 * D_);
    vq_kernel<<<dim3(B_ * 30), dim3(64), 0, stream>>>(
        ws + WS_ENC, codebook, out, ws + WS_F, loss_scale);

    decoder_kernel<<<dim3(B_), dim3(256), 0, stream>>>(
        ws + WS_F, dec_w1, dec_b1, dec_w2, dec_b2, dec_w3, dec_b3, ws + WS_DEC);

    logits_wmma_kernel<<<dim3((B_ * L_) / 16), dim3(256), 0, stream>>>(
        ws + WS_DEC, out_w, out_b, out + OUT_LOGITS);
}
